// Model_26637387170048
// MI455X (gfx1250) — matrix operand order repaired and checked by reading, not compiled, not run on hardware
//
#include <hip/hip_runtime.h>
#include <cstdint>

#define NTH 256

typedef __attribute__((ext_vector_type(16))) _Float16 v16h;
typedef __attribute__((ext_vector_type(8)))  float    v8f;

// ---- problem dims ----
#define DH     2048
#define DQL    768
#define DKVL   512
#define DDR    64
#define DDN    128
#define DNH    16
#define DNIH   8
#define DIDX   128
#define DBS    16
#define DBLK   128
#define DBPS   32
#define DMAXKV (DBPS*DBLK)   // 4096
#define DTOPK  1024

// ---- workspace layout (float offsets) ----
#define WS_CQ    0                      // 16*768
#define WS_NCKV  12288                  // 16*512   (virtually-updated cache rows)
#define WS_NKR   20480                  // 16*64
#define WS_NKI   21504                  // 16*128
#define WS_QI    23552                  // 16*8*128
#define WS_HW    39936                  // 16*8
#define WS_QPE   40064                  // 16*16*64
#define WS_QLAT  56448                  // 16*16*512
#define WS_ISC   187520                 // 16*4096
#define WS_SEL   253056                 // 16*1024 (int)
#define WS_SELV  269440                 // 16*1024
#define WS_ATT   285824                 // 16*16*1024 (att -> p in place)
#define WS_OLAT  547968                 // 16*16*512

__device__ inline float block_sum(float v, float* red) {
  int tid = threadIdx.x;
  red[tid] = v; __syncthreads();
  for (int s = NTH/2; s > 0; s >>= 1) { if (tid < s) red[tid] += red[tid+s]; __syncthreads(); }
  float r = red[0]; __syncthreads();
  return r;
}

// ---------------- K1: x projections + norms + rope(kr) ----------------
__global__ void k_proj_x(const float* __restrict__ x, const float* __restrict__ w_dq,
                         const float* __restrict__ w_dkv_kr, const float* __restrict__ w_idx_k,
                         const float* __restrict__ gamma_cq, const float* __restrict__ gamma_ckv,
                         const float* __restrict__ in_gamma_k, const float* __restrict__ in_beta_k,
                         const float* __restrict__ sinv, const float* __restrict__ cosv,
                         float* __restrict__ ws) {
  __shared__ float xs[DH];
  __shared__ float cqs[DQL];
  __shared__ float cks[DKVL + DDR];
  __shared__ float kis[DIDX];
  __shared__ float red[NTH];
  int b = blockIdx.x, tid = threadIdx.x;
  for (int i = tid; i < DH; i += NTH) xs[i] = x[b*DH + i];
  __syncthreads();
  for (int j = tid; j < DQL; j += NTH) {
    float a = 0.f;
    for (int i = 0; i < DH; ++i) a += xs[i] * w_dq[i*DQL + j];
    cqs[j] = a;
  }
  for (int j = tid; j < DKVL + DDR; j += NTH) {
    float a = 0.f;
    for (int i = 0; i < DH; ++i) a += xs[i] * w_dkv_kr[i*(DKVL+DDR) + j];
    cks[j] = a;
  }
  for (int j = tid; j < DIDX; j += NTH) {
    float a = 0.f;
    for (int i = 0; i < DH; ++i) a += xs[i] * w_idx_k[i*DIDX + j];
    kis[j] = a;
  }
  __syncthreads();
  float p = 0.f;
  for (int j = tid; j < DQL; j += NTH) p += cqs[j]*cqs[j];
  float ss_cq = block_sum(p, red);
  p = 0.f;
  for (int j = tid; j < DKVL; j += NTH) p += cks[j]*cks[j];
  float ss_ckv = block_sum(p, red);
  p = 0.f;
  for (int j = tid; j < DIDX; j += NTH) p += kis[j];
  float s_ki = block_sum(p, red);
  p = 0.f;
  for (int j = tid; j < DIDX; j += NTH) p += kis[j]*kis[j];
  float ss_ki = block_sum(p, red);
  float r_cq  = rsqrtf(ss_cq/DQL + 1e-6f);
  float r_ckv = rsqrtf(ss_ckv/DKVL + 1e-6f);
  float mki   = s_ki/DIDX;
  float r_ki  = rsqrtf(ss_ki/DIDX - mki*mki + 1e-6f);
  for (int j = tid; j < DQL;  j += NTH) ws[WS_CQ   + b*DQL  + j] = cqs[j]*r_cq*gamma_cq[j];
  for (int j = tid; j < DKVL; j += NTH) ws[WS_NCKV + b*DKVL + j] = cks[j]*r_ckv*gamma_ckv[j];
  for (int j = tid; j < DIDX; j += NTH)
    ws[WS_NKI + b*DIDX + j] = (kis[j]-mki)*r_ki*in_gamma_k[j] + in_beta_k[j];
  for (int j = tid; j < DDR; j += NTH) {
    float v   = cks[DKVL + j];
    float rot = (j < DDR/2) ? -cks[DKVL + j + DDR/2] : cks[DKVL + j - DDR/2];
    ws[WS_NKR + b*DDR + j] = v*cosv[b*DDR + j] + rot*sinv[b*DDR + j];
  }
}

// ---------------- K2: cq projections, rope(q_pe), q_lat ----------------
__global__ void k_proj_q(const float* __restrict__ w_uq_qr, const float* __restrict__ w_idx_qb,
                         const float* __restrict__ w_idx_proj, const float* __restrict__ w_uk,
                         const float* __restrict__ sinv, const float* __restrict__ cosv,
                         float* __restrict__ ws) {
  __shared__ float cqs[DQL];
  __shared__ float qn[DNH*DDN];
  __shared__ float pe[DNH*DDR];
  int b = blockIdx.x, tid = threadIdx.x;
  for (int i = tid; i < DQL; i += NTH) cqs[i] = ws[WS_CQ + b*DQL + i];
  __syncthreads();
  const int QW = DNH*(DDN+DDR); // 3072
  for (int j = tid; j < QW; j += NTH) {
    float a = 0.f;
    for (int i = 0; i < DQL; ++i) a += cqs[i] * w_uq_qr[i*QW + j];
    int h = j / (DDN+DDR), d = j % (DDN+DDR);
    if (d < DDN) qn[h*DDN + d] = a; else pe[h*DDR + (d-DDN)] = a;
  }
  for (int j = tid; j < DNIH*DIDX; j += NTH) {
    float a = 0.f;
    for (int i = 0; i < DQL; ++i) a += cqs[i] * w_idx_qb[i*(DNIH*DIDX) + j];
    ws[WS_QI + b*DNIH*DIDX + j] = a;
  }
  if (tid < DNIH) {
    float a = 0.f;
    for (int i = 0; i < DQL; ++i) a += cqs[i] * w_idx_proj[i*DNIH + tid];
    ws[WS_HW + b*DNIH + tid] = a;
  }
  __syncthreads();
  for (int j = tid; j < DNH*DDR; j += NTH) {
    int h = j / DDR, r = j % DDR;
    float v   = pe[j];
    float rot = (r < DDR/2) ? -pe[h*DDR + r + DDR/2] : pe[h*DDR + r - DDR/2];
    ws[WS_QPE + b*DNH*DDR + j] = v*cosv[b*DDR + r] + rot*sinv[b*DDR + r];
  }
  for (int j = tid; j < DNH*DKVL; j += NTH) {
    int h = j / DKVL, c = j % DKVL;
    float a = 0.f;
    for (int d = 0; d < DDN; ++d) a += qn[h*DDN + d] * w_uk[(h*DDN + d)*DKVL + c];
    ws[WS_QLAT + b*DNH*DKVL + j] = a;
  }
}

// ---------------- K3: index scoring (WMMA f16, fp32 acc) ----------------
__global__ void k_index_score(const float* __restrict__ index_k_cache,
                              const int* __restrict__ block_table,
                              const int* __restrict__ cache_index,
                              const int* __restrict__ act_seqs,
                              float* __restrict__ ws) {
  int b = blockIdx.y, kb = blockIdx.x;
  int tid = threadIdx.x;
  int wave = tid >> 5, lane = tid & 31;
  int m = lane & 15, g = lane >> 4;
  const float* qi = ws + WS_QI + b*DNIH*DIDX;
  v16h afrag[4];
  #pragma unroll
  for (int ks = 0; ks < 4; ++ks) {
    #pragma unroll
    for (int e = 0; e < 16; ++e) {
      int k = ks*32 + 16*g + e;
      afrag[ks][e] = (m < DNIH) ? (_Float16)qi[m*DIDX + k] : (_Float16)0.0f;
    }
  }
  float hw[8];
  #pragma unroll
  for (int r = 0; r < 8; ++r) hw[r] = ws[WS_HW + b*DNIH + r];
  int nb   = block_table[b*DBPS + kb];
  int nact = act_seqs[b];
  for (int t = 0; t < 2; ++t) {
    int off = (wave*2 + t)*16 + m;     // key offset in this cache block (B column)
    int pos = nb*DBLK + off;           // flat cache row == pos value
    const float* kptr = index_k_cache + (size_t)pos*DIDX;
    for (int bb = 0; bb < DBS; ++bb)
      if (pos == cache_index[bb]) kptr = ws + WS_NKI + bb*DIDX;  // virtual cache update
    v8f acc = {};
    #pragma unroll
    for (int ks = 0; ks < 4; ++ks) {
      v16h bfrag;
      const float* src = kptr + ks*32 + 16*g;
      #pragma unroll
      for (int e = 0; e < 16; ++e) bfrag[e] = (_Float16)src[e];
      acc = __builtin_amdgcn_wmma_f32_16x16x32_f16(false, afrag[ks], false, bfrag,
                                                   (short)0, acc, false, false);
    }
    if (g == 0) {  // rows 0..7 = the 8 real index heads
      float s = 0.f;
      #pragma unroll
      for (int r = 0; r < 8; ++r) { float v = acc[r]; s += hw[r]*(v > 0.f ? v : 0.f); }
      s *= 0.08838834764831845f;  // 1/sqrt(IDX)
      int n = kb*DBLK + off;
      ws[WS_ISC + b*DMAXKV + n] = (n < nact) ? s : -1e9f;
    }
  }
}

// ---------------- K4: deterministic radix-select top-1024 ----------------
__device__ inline unsigned f2ord(float f) {
  unsigned u = __float_as_uint(f);
  return (u & 0x80000000u) ? ~u : (u | 0x80000000u);
}

__global__ void k_topk(const int* __restrict__ block_table, float* __restrict__ ws) {
  __shared__ unsigned hist[256];
  __shared__ unsigned sbuf[NTH];
  __shared__ unsigned sh_prefix, sh_k;
  int b = blockIdx.x, tid = threadIdx.x;
  const float* sc = ws + WS_ISC + b*DMAXKV;
  int*   sel  = (int*)(ws + WS_SEL) + b*DTOPK;
  float* selv = ws + WS_SELV + b*DTOPK;
  if (tid == 0) { sh_prefix = 0u; sh_k = DTOPK; }
  __syncthreads();
  for (int byte = 3; byte >= 0; --byte) {
    hist[tid] = 0u;
    __syncthreads();
    unsigned pfx = sh_prefix;
    unsigned hm = (byte == 3) ? 0u : (0xFFFFFFFFu << ((byte+1)*8));
    for (int n = tid; n < DMAXKV; n += NTH) {
      unsigned u = f2ord(sc[n]);
      if ((u & hm) == (pfx & hm)) atomicAdd(&hist[(u >> (byte*8)) & 255u], 1u);
    }
    __syncthreads();
    if (tid == 0) {
      unsigned k = sh_k, cum = 0u; int bs2 = 0;
      for (int i = 255; i >= 0; --i) {
        if (cum + hist[i] >= k) { bs2 = i; break; }
        cum += hist[i];
      }
      sh_prefix = pfx | ((unsigned)bs2 << (byte*8));
      sh_k = k - cum;
    }
    __syncthreads();
  }
  unsigned T = sh_prefix;       // exact key of the 1024th-largest
  unsigned R = sh_k;            // ties of T to take (lowest index first)
  unsigned Cgt = DTOPK - R;     // strictly-greater count
  int cg = 0, ce = 0;
  for (int i = 0; i < 16; ++i) {
    unsigned u = f2ord(sc[tid*16 + i]);
    cg += (u > T); ce += (u == T);
  }
  sbuf[tid] = (unsigned)cg; __syncthreads();
  for (int off = 1; off < NTH; off <<= 1) {
    unsigned v = (tid >= off) ? sbuf[tid - off] : 0u; __syncthreads();
    sbuf[tid] += v; __syncthreads();
  }
  unsigned gbase = sbuf[tid] - (unsigned)cg;
  __syncthreads();
  sbuf[tid] = (unsigned)ce; __syncthreads();
  for (int off = 1; off < NTH; off <<= 1) {
    unsigned v = (tid >= off) ? sbuf[tid - off] : 0u; __syncthreads();
    sbuf[tid] += v; __syncthreads();
  }
  unsigned ebase = sbuf[tid] - (unsigned)ce;
  __syncthreads();
  int lg = 0, le = 0;
  for (int i = 0; i < 16; ++i) {
    int n = tid*16 + i;
    unsigned u = f2ord(sc[n]);
    int slot = -1;
    if (u > T) slot = (int)(gbase + (unsigned)lg++);
    else if (u == T) { unsigned er = ebase + (unsigned)le++; if (er < R) slot = (int)(Cgt + er); }
    if (slot >= 0) {
      sel[slot]  = block_table[b*DBPS + (n >> 7)]*DBLK + (n & (DBLK-1));
      selv[slot] = sc[n];
    }
  }
}

// ---------------- K5: attention logits over selected keys (WMMA) ----------------
__global__ void k_att(const float* __restrict__ kv_cache, const float* __restrict__ kr_cache,
                      const int* __restrict__ cache_index, float* __restrict__ ws) {
  __shared__ _Float16 qh[DNH][DKVL + DDR];  // 16 x 576 f16
  int b = blockIdx.x, tid = threadIdx.x;
  int wave = tid >> 5, lane = tid & 31;
  int m = lane & 15, g = lane >> 4;
  for (int j = tid; j < DNH*(DKVL+DDR); j += NTH) {
    int h = j / (DKVL+DDR), c = j % (DKVL+DDR);
    float v = (c < DKVL) ? ws[WS_QLAT + (b*DNH + h)*DKVL + c]
                         : ws[WS_QPE  + (b*DNH + h)*DDR  + (c - DKVL)];
    qh[h][c] = (_Float16)v;
  }
  __syncthreads();
  const int* sel = (const int*)(ws + WS_SEL);
  const float scale = 0.07216878364870323f; // 1/sqrt(DN+DR)
  for (int t = wave; t < DTOPK/16; t += NTH/32) {
    int slot = t*16 + m;
    int row  = sel[b*DTOPK + slot];
    float sval = ws[WS_SELV + b*DTOPK + slot];
    const float* ckvp = kv_cache + (size_t)row*DKVL;
    const float* krp  = kr_cache + (size_t)row*DDR;
    for (int bb = 0; bb < DBS; ++bb)
      if (row == cache_index[bb]) { ckvp = ws + WS_NCKV + bb*DKVL; krp = ws + WS_NKR + bb*DDR; }
    {   // prefetch the rows this lane will gather on its next tile
      int tn = t + NTH/32;
      if (tn < DTOPK/16) {
        int nrow = sel[b*DTOPK + tn*16 + m];
        __builtin_prefetch(kv_cache + (size_t)nrow*DKVL, 0, 1);
        __builtin_prefetch(kr_cache + (size_t)nrow*DDR, 0, 1);
      }
    }
    v8f acc = {};
    for (int ks = 0; ks < (DKVL+DDR)/32; ++ks) {
      int k0 = ks*32;
      v16h afrag, bfrag;
      #pragma unroll
      for (int e = 0; e < 16; ++e) {
        int k = k0 + 16*g + e;
        afrag[e] = qh[m][k];
      }
      const float* src = (k0 < DKVL) ? (ckvp + k0 + 16*g) : (krp + (k0 - DKVL) + 16*g);
      #pragma unroll
      for (int e = 0; e < 16; ++e) bfrag[e] = (_Float16)src[e];
      acc = __builtin_amdgcn_wmma_f32_16x16x32_f16(false, afrag, false, bfrag,
                                                   (short)0, acc, false, false);
    }
    bool valid = sval > -1e8f;
    #pragma unroll
    for (int r = 0; r < 8; ++r) {
      int h = r + 8*g;
      ws[WS_ATT + (b*DNH + h)*DTOPK + slot] = valid ? acc[r]*scale : -1e9f;
    }
  }
}

// ---------------- K6: softmax in place ----------------
__global__ void k_softmax(float* __restrict__ att) {
  __shared__ float red[NTH];
  int row = blockIdx.x, tid = threadIdx.x;
  float* a = att + (size_t)row*DTOPK;
  float mx = -1e30f;
  for (int i = tid; i < DTOPK; i += NTH) mx = fmaxf(mx, a[i]);
  red[tid] = mx; __syncthreads();
  for (int s = NTH/2; s > 0; s >>= 1) { if (tid < s) red[tid] = fmaxf(red[tid], red[tid+s]); __syncthreads(); }
  mx = red[0]; __syncthreads();
  float sum = 0.f;
  for (int i = tid; i < DTOPK; i += NTH) { float e = __expf(a[i] - mx); a[i] = e; sum += e; }
  red[tid] = sum; __syncthreads();
  for (int s = NTH/2; s > 0; s >>= 1) { if (tid < s) red[tid] += red[tid+s]; __syncthreads(); }
  float inv = 1.f / red[0];
  for (int i = tid; i < DTOPK; i += NTH) a[i] *= inv;
}

// ---------------- K7: o_lat = P @ ckv_sel (WMMA; async global->LDS staging) ----------------
__global__ void k_olat(const float* __restrict__ kv_cache, const int* __restrict__ cache_index,
                       float* __restrict__ ws) {
  __shared__ float ksh[32][DKVL];  // 32 keys x 512 ch f32 (64KB), filled by async DMA
  int b = blockIdx.x, tid = threadIdx.x;
  int wave = tid >> 5, lane = tid & 31;
  int m = lane & 15, g = lane >> 4;
  const int* sel = (const int*)(ws + WS_SEL);
  int r  = tid >> 3;            // 8 threads per key row
  int c0 = (tid & 7) * 64;      // 64 floats (4 x b128) per thread
  unsigned lds0 = (unsigned)(uintptr_t)&ksh[r][c0];  // low 32 bits of generic addr = LDS offset
  v8f acc[4] = {};
  for (int kc = 0; kc < DTOPK/32; ++kc) {
    int row = sel[b*DTOPK + kc*32 + r];
    const float* src = kv_cache + (size_t)row*DKVL;
    for (int bb = 0; bb < DBS; ++bb)
      if (row == cache_index[bb]) src = ws + WS_NCKV + bb*DKVL;  // virtual cache update
    src += c0;
    // asynchronous gather: global -> LDS, ASYNCcnt-tracked, no VGPR staging
    #pragma unroll
    for (int q = 0; q < 4; ++q)
      asm volatile("global_load_async_to_lds_b128 %0, %1, off"
                   :: "v"(lds0 + q*64), "v"(src + q*16) : "memory");
    if (kc + 1 < DTOPK/32) {    // prefetch next chunk's gathered row into L2
      int nrow = sel[b*DTOPK + (kc+1)*32 + r];
      __builtin_prefetch(kv_cache + (size_t)nrow*DKVL + c0, 0, 1);
    }
    asm volatile("s_wait_asynccnt 0" ::: "memory");
    __syncthreads();
    v16h afrag;   // P tile: 16 heads x 32 keys (read directly; L2-resident)
    #pragma unroll
    for (int e = 0; e < 16; ++e) {
      int k = 16*g + e;
      afrag[e] = (_Float16)ws[WS_ATT + (b*DNH + m)*DTOPK + kc*32 + k];
    }
    #pragma unroll
    for (int i = 0; i < 4; ++i) {
      int ch = (wave*4 + i)*16 + m;
      v16h bfrag;
      #pragma unroll
      for (int e = 0; e < 16; ++e) bfrag[e] = (_Float16)ksh[16*g + e][ch];
      acc[i] = __builtin_amdgcn_wmma_f32_16x16x32_f16(false, afrag, false, bfrag,
                                                      (short)0, acc[i], false, false);
    }
    __syncthreads();
  }
  #pragma unroll
  for (int i = 0; i < 4; ++i) {
    int ch = (wave*4 + i)*16 + m;
    #pragma unroll
    for (int rr = 0; rr < 8; ++rr) {
      int h = rr + 8*g;
      ws[WS_OLAT + (b*DNH + h)*DKVL + ch] = acc[i][rr];
    }
  }
}

// ---------------- K8: o = o_lat @ w_uk^T ----------------
__global__ void k_out(const float* __restrict__ w_uk, const float* __restrict__ ws,
                      float* __restrict__ out) {
  __shared__ float ol[DNH*DKVL];
  int b = blockIdx.x, tid = threadIdx.x;
  for (int j = tid; j < DNH*DKVL; j += NTH) ol[j] = ws[WS_OLAT + b*DNH*DKVL + j];
  __syncthreads();
  for (int j = tid; j < DNH*DDN; j += NTH) {
    int h = j / DDN, d = j % DDN;
    float a = 0.f;
    for (int c = 0; c < DKVL; ++c) a += ol[h*DKVL + c] * w_uk[(h*DDN + d)*DKVL + c];
    out[b*DNH*DDN + j] = a;
  }
}

extern "C" void kernel_launch(void* const* d_in, const int* in_sizes, int n_in,
                              void* d_out, int out_size, void* d_ws, size_t ws_size,
                              hipStream_t stream) {
  (void)in_sizes; (void)n_in; (void)out_size; (void)ws_size;
  const float* x           = (const float*)d_in[0];
  const float* w_dq        = (const float*)d_in[1];
  const float* w_uq_qr     = (const float*)d_in[2];
  const float* w_uk        = (const float*)d_in[3];
  const float* w_dkv_kr    = (const float*)d_in[4];
  const float* gamma_cq    = (const float*)d_in[5];
  const float* gamma_ckv   = (const float*)d_in[6];
  const float* sinv        = (const float*)d_in[7];
  const float* cosv        = (const float*)d_in[8];
  const int*   cache_index = (const int*)d_in[9];
  const float* kv_cache    = (const float*)d_in[10];
  const float* kr_cache    = (const float*)d_in[11];
  const int*   block_table = (const int*)d_in[12];
  const int*   act_seqs    = (const int*)d_in[13];
  const float* w_idx_qb    = (const float*)d_in[14];
  const float* w_idx_k     = (const float*)d_in[15];
  const float* w_idx_proj  = (const float*)d_in[16];
  const float* in_gamma_k  = (const float*)d_in[17];
  const float* in_beta_k   = (const float*)d_in[18];
  const float* index_k_cache = (const float*)d_in[19];
  float* ws  = (float*)d_ws;
  float* out = (float*)d_out;

  k_proj_x<<<DBS, NTH, 0, stream>>>(x, w_dq, w_dkv_kr, w_idx_k, gamma_cq, gamma_ckv,
                                    in_gamma_k, in_beta_k, sinv, cosv, ws);
  k_proj_q<<<DBS, NTH, 0, stream>>>(w_uq_qr, w_idx_qb, w_idx_proj, w_uk, sinv, cosv, ws);
  k_index_score<<<dim3(DBPS, DBS), 128, 0, stream>>>(index_k_cache, block_table,
                                                     cache_index, act_seqs, ws);
  k_topk<<<DBS, NTH, 0, stream>>>(block_table, ws);
  k_att<<<DBS, NTH, 0, stream>>>(kv_cache, kr_cache, cache_index, ws);
  k_softmax<<<DBS*DNH, NTH, 0, stream>>>(ws + WS_ATT);
  k_olat<<<DBS, NTH, 0, stream>>>(kv_cache, cache_index, ws);
  k_out<<<DBS, NTH, 0, stream>>>(w_uk, ws, out);
}
